// Raymarcher_11252814316135
// MI455X (gfx1250) — matrix operand order repaired and checked by reading, not compiled, not run on hardware
//
#include <hip/hip_runtime.h>
#include <hip/hip_bf16.h>
#include <math.h>

typedef __attribute__((ext_vector_type(16))) _Float16 v16h;
typedef __attribute__((ext_vector_type(8)))  float    v8f;

#define BN 65536      // B*N
#define MT 64         // points per workgroup
#define AS 264        // LDS half-stride for 256-wide activation tiles
#define GS 68         // LDS float-stride for 64-wide gate tile

// d_out layout (floats): wc_final[BN*3] | last_depth[BN] | all_depth[9*BN]
#define OUT_LAST  196608
#define OUT_DEPTH 262144

__device__ __forceinline__ float sigmf(float x) { return 1.0f / (1.0f + __expf(-x)); }

// ---------------- prep: depth row of inv(c2w) (affine inverse) -------------
__global__ void prep_inv_kernel(const float* __restrict__ c2w, float* __restrict__ invrow) {
    int b = threadIdx.x;
    if (b >= 8) return;
    const float* M = c2w + b * 16;
    float a00=M[0], a01=M[1], a02=M[2],  t0=M[3];
    float a10=M[4], a11=M[5], a12=M[6],  t1=M[7];
    float a20=M[8], a21=M[9], a22=M[10], t2=M[11];
    float det = a00*(a11*a22 - a12*a21) - a01*(a10*a22 - a12*a20) + a02*(a10*a21 - a11*a20);
    float id = 1.0f / det;
    float r0 = (a10*a21 - a11*a20) * id;
    float r1 = (a01*a20 - a00*a21) * id;
    float r2 = (a00*a11 - a01*a10) * id;
    invrow[b*4+0] = r0; invrow[b*4+1] = r1; invrow[b*4+2] = r2;
    invrow[b*4+3] = -(r0*t0 + r1*t1 + r2*t2);
}

// ---------------- prep: ray dirs, wc0, initial depth out -------------------
__global__ __launch_bounds__(256) void prep_points_kernel(
    const float* __restrict__ c2w, const float* __restrict__ uv,
    const float* __restrict__ intr, const float* __restrict__ d0,
    float* __restrict__ rd, float* __restrict__ wc0, float* __restrict__ out) {
    int idx = blockIdx.x * 256 + threadIdx.x;
    if (idx >= BN) return;
    int b = idx >> 13;
    const float* K = intr + b * 9;
    float fx = K[0], sk = K[1], cx = K[2], fy = K[4], cy = K[5];
    const float* M = c2w + b * 16;
    float x = uv[idx*2+0], y = uv[idx*2+1];
    float lx = (x - cx + cy*sk/fy - sk*y/fy) / fx;   // per-unit-depth lift
    float ly = (y - cy) / fy;
    // world point at z = 1
    float wx = M[0]*lx + M[1]*ly + M[2]  + M[3];
    float wy = M[4]*lx + M[5]*ly + M[6]  + M[7];
    float wz = M[8]*lx + M[9]*ly + M[10] + M[11];
    float dx = wx - M[3], dy = wy - M[7], dz = wz - M[11];
    float rn = 1.0f / sqrtf(dx*dx + dy*dy + dz*dz);
    rd[idx*3+0] = dx*rn; rd[idx*3+1] = dy*rn; rd[idx*3+2] = dz*rn;
    float z0 = d0[idx];
    wc0[idx*3+0] = M[0]*lx*z0 + M[1]*ly*z0 + M[2]*z0  + M[3];
    wc0[idx*3+1] = M[4]*lx*z0 + M[5]*ly*z0 + M[6]*z0  + M[7];
    wc0[idx*3+2] = M[8]*lx*z0 + M[9]*ly*z0 + M[10]*z0 + M[11];
    out[OUT_DEPTH + idx] = z0;                       // all_depth[0]
}

// ------- prep: f16 fragment-swizzled B matrices (phi_w2, w_ih^T) -----------
// packed index p = ((ncol_tile*8 + k0)*32 + lane)*16 + i
// lane 0-15: N = tile*16+lane, K = k0*32 + i ; lane 16-31: K = k0*32 + 16 + i
__global__ __launch_bounds__(256) void prep_weights_kernel(
    const float* __restrict__ phi_w2, const float* __restrict__ w_ih,
    _Float16* __restrict__ w2p, _Float16* __restrict__ wip) {
    int p = blockIdx.x * 256 + threadIdx.x;
    if (p < 65536) {
        int i = p & 15, lane = (p >> 4) & 31, k0 = (p >> 9) & 7, ni = p >> 12;
        int n = ni * 16 + (lane & 15);
        int k = k0 * 32 + ((lane & 16) ? 16 : 0) + i;
        w2p[p] = (_Float16)phi_w2[k * 256 + n];      // B[k][n] = phi_w2[k][n]
    }
    if (p < 16384) {
        int i = p & 15, lane = (p >> 4) & 31, k0 = (p >> 9) & 7, nj = p >> 12;
        int n = nj * 16 + (lane & 15);
        int k = k0 * 32 + ((lane & 16) ? 16 : 0) + i;
        wip[p] = (_Float16)w_ih[n * 256 + k];        // B[k][j] = w_ih[j][k]
    }
}

// ---------------------------- fused march ----------------------------------
__global__ __launch_bounds__(256) void march_kernel(
    const float* __restrict__ rd_g, const float* __restrict__ wc0_g,
    const float* __restrict__ invrow,
    const _Float16* __restrict__ w2p, const _Float16* __restrict__ wip,
    const float* __restrict__ phi_w1, const float* __restrict__ phi_b1,
    const float* __restrict__ phi_b2,
    const float* __restrict__ w_hh, const float* __restrict__ b_ih,
    const float* __restrict__ b_hh, const float* __restrict__ w_out,
    const float* __restrict__ b_out, float* __restrict__ out) {

    // ---- LDS (~262 KB total; CDNA5 allows up to 320 KB per workgroup) ----
    __shared__ _Float16 sW2[65536];   // 128 KB swizzled phi_w2 B-fragments
    __shared__ _Float16 sWI[16384];   //  32 KB swizzled w_ih^T B-fragments
    __shared__ _Float16 sA[MT * AS];
    __shared__ _Float16 sX[MT * AS];
    __shared__ float sG[MT * GS];
    __shared__ float sWC[MT * 3];
    __shared__ float sRD[MT * 3];
    __shared__ float sH[MT * 16];
    __shared__ float sC[MT * 16];
    __shared__ float sW1[768];
    __shared__ float sB1[256];
    __shared__ float sB2[256];
    __shared__ float sWhh[1024];
    __shared__ float sBih[64];
    __shared__ float sBhh[64];
    __shared__ float sWout[16];
    __shared__ float sInv[4];
    __shared__ float sBout;

    const int tid  = threadIdx.x;
    const int base = blockIdx.x * MT;
    const int gb   = base >> 13;          // batch index (8192 pts per batch)
    const int wave = tid >> 5;
    const int lane = tid & 31;
    const int mrow = lane & 15;
    const int kb   = (lane & 16) ? 8 : 0; // A-fragment K sub-offset (x2 => contiguous k = 16h + s)
    const int dmo  = (lane & 16) ? 8 : 0; // D-fragment M offset

    // Stage weights into LDS (written in-kernel => loads below can't be
    // hoisted out of the step loop and spilled, unlike the global version).
    {
        const uint4* g2 = (const uint4*)w2p;
        uint4*       l2 = (uint4*)sW2;
        for (int i = tid; i < 8192; i += 256) l2[i] = g2[i];
        const uint4* gi = (const uint4*)wip;
        uint4*       li = (uint4*)sWI;
        for (int i = tid; i < 2048; i += 256) li[i] = gi[i];
    }
    for (int i = tid; i < 768; i += 256) sW1[i] = phi_w1[i];
    if (tid < 256) { sB1[tid] = phi_b1[tid]; sB2[tid] = phi_b2[tid]; }
    for (int i = tid; i < 1024; i += 256) sWhh[i] = w_hh[i];
    if (tid < 64) { sBih[tid] = b_ih[tid]; sBhh[tid] = b_hh[tid]; }
    if (tid < 16) sWout[tid] = w_out[tid];
    if (tid < 4)  sInv[tid] = invrow[gb*4 + tid];
    if (tid == 0) sBout = b_out[0];
    for (int i = tid; i < MT*3; i += 256) { sWC[i] = wc0_g[base*3 + i]; sRD[i] = rd_g[base*3 + i]; }
    for (int i = tid; i < MT*16; i += 256) { sH[i] = 0.f; sC[i] = 0.f; }
    __syncthreads();

    for (int s = 0; s < 8; ++s) {
        // ---- layer1 (K=3, VALU): relu(wc @ phi_w1 + b1) -> sA (f16) ----
        for (int e = tid; e < MT*256; e += 256) {
            int m = e >> 8, n = e & 255;
            float v = sB1[n] + sWC[m*3+0]*sW1[n] + sWC[m*3+1]*sW1[256+n] + sWC[m*3+2]*sW1[512+n];
            sA[m*AS + n] = (_Float16)fmaxf(v, 0.0f);
        }
        __syncthreads();

        // ---- GEMM1: sX = sA(64x256) @ phi_w2(256x256) + b2  [WMMA] ----
        for (int t = 0; t < 8; ++t) {
            int tile = wave*8 + t, mi = tile >> 4, ni = tile & 15;
            v8f acc = {0.f,0.f,0.f,0.f,0.f,0.f,0.f,0.f};
            const int ml = mi*16 + mrow;
            for (int k0 = 0; k0 < 8; ++k0) {
                v16h a, bfrag;
                const _Float16* ap = &sA[ml*AS + k0*32];
                #pragma unroll
                for (int i = 0; i < 8; ++i) { a[i] = ap[2*kb + i]; a[8+i] = ap[2*kb + 8 + i]; }
                bfrag = *(const v16h*)(sW2 + ((((ni*8 + k0)*32) + lane) << 4));
                acc = __builtin_amdgcn_wmma_f32_16x16x32_f16(false, a, false, bfrag,
                                                             (short)0, acc, false, false);
            }
            int n = ni*16 + (lane & 15);
            int mb = mi*16 + dmo;
            #pragma unroll
            for (int r = 0; r < 8; ++r) sX[(mb + r)*AS + n] = (_Float16)(acc[r] + sB2[n]);
        }
        __syncthreads();

        // ---- GEMM2: sG = sX(64x256) @ w_ih^T(256x64) + b_ih  [WMMA] ----
        for (int t = 0; t < 2; ++t) {
            int tile = wave*2 + t, mi = tile >> 2, nj = tile & 3;
            v8f acc = {0.f,0.f,0.f,0.f,0.f,0.f,0.f,0.f};
            const int ml = mi*16 + mrow;
            for (int k0 = 0; k0 < 8; ++k0) {
                v16h a, bfrag;
                const _Float16* ap = &sX[ml*AS + k0*32];
                #pragma unroll
                for (int i = 0; i < 8; ++i) { a[i] = ap[2*kb + i]; a[8+i] = ap[2*kb + 8 + i]; }
                bfrag = *(const v16h*)(sWI + ((((nj*8 + k0)*32) + lane) << 4));
                acc = __builtin_amdgcn_wmma_f32_16x16x32_f16(false, a, false, bfrag,
                                                             (short)0, acc, false, false);
            }
            int n = nj*16 + (lane & 15);
            int mb = mi*16 + dmo;
            #pragma unroll
            for (int r = 0; r < 8; ++r) sG[(mb + r)*GS + n] = acc[r] + sBih[n];
        }
        __syncthreads();

        // ---- LSTM cell (H=16): gates += h @ w_hh^T + b_hh; update h,c ----
        float hn[4], cn[4];
        #pragma unroll
        for (int rep = 0; rep < 4; ++rep) {
            int pair = rep*256 + tid, m = pair >> 4, k = pair & 15;
            float gi = sG[m*GS +      k] + sBhh[k];
            float gf = sG[m*GS + 16 + k] + sBhh[16 + k];
            float gg = sG[m*GS + 32 + k] + sBhh[32 + k];
            float go = sG[m*GS + 48 + k] + sBhh[48 + k];
            #pragma unroll
            for (int kk = 0; kk < 16; ++kk) {
                float hv = sH[m*16 + kk];
                gi += hv * sWhh[(     k)*16 + kk];
                gf += hv * sWhh[(16 + k)*16 + kk];
                gg += hv * sWhh[(32 + k)*16 + kk];
                go += hv * sWhh[(48 + k)*16 + kk];
            }
            float cc = sigmf(gf) * sC[m*16 + k] + sigmf(gi) * tanhf(gg);
            cn[rep] = cc;
            hn[rep] = sigmf(go) * tanhf(cc);
        }
        __syncthreads();
        #pragma unroll
        for (int rep = 0; rep < 4; ++rep) {
            int pair = rep*256 + tid, m = pair >> 4, k = pair & 15;
            sH[m*16 + k] = hn[rep];
            sC[m*16 + k] = cn[rep];
        }
        __syncthreads();

        // ---- depth head + ray advance + depth write ----
        if (tid < MT) {
            int m = tid;
            float sd = sBout;
            #pragma unroll
            for (int k = 0; k < 16; ++k) sd += sH[m*16 + k] * sWout[k];
            float x = sWC[m*3+0] + sRD[m*3+0]*sd;
            float y = sWC[m*3+1] + sRD[m*3+1]*sd;
            float z = sWC[m*3+2] + sRD[m*3+2]*sd;
            sWC[m*3+0] = x; sWC[m*3+1] = y; sWC[m*3+2] = z;
            float depth = sInv[0]*x + sInv[1]*y + sInv[2]*z + sInv[3];
            int gidx = base + m;
            out[OUT_DEPTH + (s + 1)*BN + gidx] = depth;
            if (s == 7) {
                out[gidx*3+0] = x; out[gidx*3+1] = y; out[gidx*3+2] = z;
                out[OUT_LAST + gidx] = depth;
            }
        }
        __syncthreads();
    }
}

extern "C" void kernel_launch(void* const* d_in, const int* in_sizes, int n_in,
                              void* d_out, int out_size, void* d_ws, size_t ws_size,
                              hipStream_t stream) {
    const float* c2w    = (const float*)d_in[0];
    const float* uv     = (const float*)d_in[1];
    const float* intr   = (const float*)d_in[2];
    const float* d0     = (const float*)d_in[3];
    const float* phi_w1 = (const float*)d_in[4];
    const float* phi_b1 = (const float*)d_in[5];
    const float* phi_w2 = (const float*)d_in[6];
    const float* phi_b2 = (const float*)d_in[7];
    const float* w_ih   = (const float*)d_in[8];
    const float* w_hh   = (const float*)d_in[9];
    const float* b_ih   = (const float*)d_in[10];
    const float* b_hh   = (const float*)d_in[11];
    const float* w_out  = (const float*)d_in[12];
    const float* b_out  = (const float*)d_in[13];
    float* out = (float*)d_out;

    float*    rd     = (float*)d_ws;          // BN*3
    float*    wc0    = rd + BN*3;             // BN*3
    float*    invrow = wc0 + BN*3;            // 8*4
    _Float16* w2p    = (_Float16*)(invrow + 32);   // 65536 halves
    _Float16* wip    = w2p + 65536;                // 16384 halves

    prep_inv_kernel    <<<1,   32, 0, stream>>>(c2w, invrow);
    prep_points_kernel <<<256, 256, 0, stream>>>(c2w, uv, intr, d0, rd, wc0, out);
    prep_weights_kernel<<<256, 256, 0, stream>>>(phi_w2, w_ih, w2p, wip);
    march_kernel<<<BN / MT, 256, 0, stream>>>(rd, wc0, invrow, w2p, wip,
                                              phi_w1, phi_b1, phi_b2,
                                              w_hh, b_ih, b_hh, w_out, b_out, out);
}
